// MyLayer_64287070486580
// MI455X (gfx1250) — compile-verified, matrix operand order checked by reading, not run on hardware
//
#include <hip/hip_runtime.h>

// ---------------------------------------------------------------------------
// Per-channel dot product: out[c] = sum_{p<512*512} w[p,c] * x[p,c], c<256.
// Memory-bound (512 MB streamed, ~0.25 FLOP/B -> ~22us at 23.3 TB/s).
// Phase 1: NT b128 streaming FMA -> per-block partial rows in d_ws.
// Phase 2: column reduction on the matrix pipe via v_wmma_f32_16x16x32_f16
//          with A = all-ones; f32 partials are fed as (hi,lo) f16 Dekker
//          pairs, so hi+lo carries ~22 mantissa bits and the f32 accumulator
//          restores effectively-f32 accuracy. With A = ones, any K-slot
//          permutation of B sums identically, so no layout risk.
// Deterministic: fixed-order sums, no float atomics.
// ---------------------------------------------------------------------------

typedef __attribute__((ext_vector_type(4)))  float    v4f;
typedef __attribute__((ext_vector_type(8)))  float    v8f;
typedef __attribute__((ext_vector_type(16))) _Float16 v16h;

#define CH        256                // channels (contiguous, fastest dim)
#define POS       (512 * 512)        // spatial positions in batch-0 image
#define CHUNK_POS 256                // positions handled per block per chunk
#define NCHUNKS   (POS / CHUNK_POS)  // 1024

// ---------------------------------------------------------------------------
// Phase 1: streaming FMA. Block = 256 threads: quad = tid&63 owns channels
// [4*quad, 4*quad+4); pg = tid>>6 owns positions pg, pg+4, ... within a chunk.
// Each wave (32 lanes, same pg) issues one contiguous 512B b128 load per
// operand per position -> fully coalesced. NT hint: data is touched once.
// ---------------------------------------------------------------------------
__global__ __launch_bounds__(256) void dot_partial_kernel(
    const float* __restrict__ x, const float* __restrict__ w,
    float* __restrict__ part, int nblocks)
{
    const int tid  = threadIdx.x;
    const int quad = tid & 63;   // float4 column within the 256-channel row
    const int pg   = tid >> 6;   // position subgroup 0..3

    v4f acc = {};
    for (int chunk = blockIdx.x; chunk < NCHUNKS; chunk += nblocks) {
        const size_t base = (size_t)chunk * CHUNK_POS * CH + 4 * quad;
        const float* xb = x + base;
        const float* wb = w + base;
        #pragma unroll 8
        for (int i = 0; i < CHUNK_POS / 4; ++i) {
            const size_t poff = (size_t)(pg + 4 * i) * CH;
            v4f xv = __builtin_nontemporal_load((const v4f*)(xb + poff));
            v4f wv = __builtin_nontemporal_load((const v4f*)(wb + poff));
            acc += xv * wv;                       // 4x v_fma_f32
        }
    }

    // Deterministic in-block reduction of the 4 position subgroups.
    __shared__ v4f sm[256];
    sm[tid] = acc;
    __syncthreads();
    if (pg == 0) {
        v4f s = (sm[quad] + sm[quad + 64]) + (sm[quad + 128] + sm[quad + 192]);
        *(v4f*)(part + (size_t)blockIdx.x * CH + 4 * quad) = s;
    }
}

// ---------------------------------------------------------------------------
// Phase 2 helper: pack 8 f32 partial rows of channel c into 16 f16 B-slots
// as (hi, lo) pairs. Branch-free OOB handling: clamp the row index so the
// load is always in-bounds (row 0 exists whenever this kernel runs), then
// mask the value with a v_cndmask. No EXEC divergence -> WMMA-safe and no
// saveexec ladders in the inner loop.
// ---------------------------------------------------------------------------
__device__ inline v16h pack_hilo(const float* __restrict__ part, int rbase,
                                 int rows, int c)
{
    v16h b;
    #pragma unroll
    for (int j = 0; j < 8; ++j) {
        const int  r    = rbase + j;
        const bool ok   = (r < rows);
        const int  rc   = ok ? r : 0;                     // in-bounds address
        float      v    = part[(size_t)rc * CH + c];      // unconditional load
        v               = ok ? v : 0.0f;                  // value mask
        const _Float16 hi = (_Float16)v;
        const _Float16 lo = (_Float16)(v - (float)hi);
        b[2 * j]     = hi;
        b[2 * j + 1] = lo;
    }
    return b;
}

// ---------------------------------------------------------------------------
// Phase 2: column-sum of part[rows][256]. D = Ones(16x32) * B(32x16) + C.
// Wave g owns channels 16g..16g+15; lane L supplies 16 of the 32 K-slots for
// channel c = 16g + (L&15): lanes 0-15 take rows rbase..rbase+7, lanes 16-31
// rows rbase+8..rbase+15 (A = ones makes slot order irrelevant). Each WMMA
// therefore reduces 16 f32 rows x 16 channels; two independent accumulator
// chains hide the D->C RAW latency. D row 0 (VGPR0, lanes 0-15) holds the
// per-channel totals.
// ---------------------------------------------------------------------------
__global__ __launch_bounds__(512) void col_reduce_wmma_kernel(
    const float* __restrict__ part, float* __restrict__ out, int rows)
{
    const int lane = threadIdx.x & 31;
    const int wid  = threadIdx.x >> 5;        // wave 0..15 -> channel group
    const int hlf  = lane >> 4;               // which 8-row slice this lane packs
    const int c    = wid * 16 + (lane & 15);  // this lane's channel column

    v16h ones;
    #pragma unroll
    for (int i = 0; i < 16; ++i) ones[i] = (_Float16)1.0f;

    v8f acc0 = {}, acc1 = {};
    for (int r = 0; r < rows; r += 32) {      // 2 WMMAs / iter, 16 rows each
        v16h b0 = pack_hilo(part, r +      8 * hlf, rows, c);
        v16h b1 = pack_hilo(part, r + 16 + 8 * hlf, rows, c);
        acc0 = __builtin_amdgcn_wmma_f32_16x16x32_f16(
                   false, ones, false, b0, (short)0, acc0, false, false);
        acc1 = __builtin_amdgcn_wmma_f32_16x16x32_f16(
                   false, ones, false, b1, (short)0, acc1, false, false);
    }

    v8f t = acc0 + acc1;                      // identical C/D layouts
    // D VGPR0, lanes 0-15: M=0, N=lane -> the per-channel column sums.
    if (hlf == 0) out[c] = t[0];
}

// ---------------------------------------------------------------------------
extern "C" void kernel_launch(void* const* d_in, const int* in_sizes, int n_in,
                              void* d_out, int out_size, void* d_ws, size_t ws_size,
                              hipStream_t stream)
{
    const float* x   = (const float*)d_in[0]; // (2,512,512,256); batch 0 only
    const float* w   = (const float*)d_in[1]; // (512,512,256)
    float*       out = (float*)d_out;         // 256 floats
    float*       prt = (float*)d_ws;          // NB x 256 partial sums

    // Clamp block count to the workspace; prefer a multiple of 32 rows so
    // the phase-2 tail guards are dead in the common path.
    int maxb = (int)(ws_size / (CH * sizeof(float)));
    int nb = NCHUNKS;                 // 1024 (needs 1 MB of d_ws)
    if (nb > maxb) nb = maxb;
    if (nb >= 32)  nb &= ~31;         // round down to multiple of 32
    if (nb < 1)    nb = 1;

    dot_partial_kernel<<<nb, 256, 0, stream>>>(x, w, prt, nb);
    col_reduce_wmma_kernel<<<1, 512, 0, stream>>>(prt, out, nb);

    (void)in_sizes; (void)n_in; (void)out_size;
}
